// Transformer_1803886264728
// MI455X (gfx1250) — matrix operand order repaired and checked by reading, not compiled, not run on hardware
//
#include <hip/hip_runtime.h>
#include <hip/hip_bf16.h>
#include <cmath>

typedef _Float16 h16;
typedef __attribute__((ext_vector_type(16))) _Float16 v16h;
typedef __attribute__((ext_vector_type(8)))  _Float16 v8h;
typedef __attribute__((ext_vector_type(8)))  float    v8f;
typedef __attribute__((ext_vector_type(4)))  float    v4f;

namespace cfg {
constexpr int B = 2, SC = 256, T = 384, D = 128, F = 512, H = 4, L = 4;
constexpr int NELEM = 50, NB = 8;
}

// ============================================================================
// WMMA GEMM:  C[i1,i2] = act( A[i1,i2] (MxK) * B[i1,i2] (KxN or NxK^T) + bias )
// 256 threads = 8 wave32; block tile 128x64; K-step 32; f32 accumulate.
// ============================================================================
template<bool TRANSB, bool RELU, bool AF16, bool OUTF16>
__global__ __launch_bounds__(256)
void gemm_wmma(const void* __restrict__ Aall, const float* __restrict__ Ball,
               const float* __restrict__ biasAll, void* __restrict__ Call,
               int M, int N, int K, int lda, int ldb, int ldc, int nb2,
               long long sA1, long long sA2, long long sB1, long long sB2,
               long long sC1, long long sC2, long long sBias1)
{
  constexpr int BM = 128, BN = 64, BK = 32, LSTR = 40; // 40 halves = 80B row (16B aligned)
  __shared__ h16 sA[BM * LSTR];
  __shared__ h16 sB[BN * LSTR];

  const int tid = threadIdx.x;
  const int z   = blockIdx.z;
  const int i1  = z / nb2, i2 = z - i1 * nb2;

  const float* Bp   = Ball + i1 * sB1 + i2 * sB2;
  const float* bias = biasAll ? (biasAll + i1 * sBias1) : nullptr;
  const long long cOff = i1 * sC1 + i2 * sC2;

  const int m0 = blockIdx.y * BM;
  const int n0 = blockIdx.x * BN;

  const int wave = tid >> 5, lane = tid & 31;
  const int wm = wave >> 1, wn = wave & 1;      // 4x2 wave grid -> 32x32 per wave
  const int l15 = lane & 15, lhi = lane >> 4;

  v8f acc[2][2];
#pragma unroll
  for (int a = 0; a < 2; ++a)
#pragma unroll
    for (int b = 0; b < 2; ++b)
#pragma unroll
      for (int e = 0; e < 8; ++e) acc[a][b][e] = 0.0f;

  const int nk = (K + BK - 1) / BK;
  for (int kt = 0; kt < nk; ++kt) {
    const int k0 = kt * BK;

    // ---- stage A tile (BM x BK) as f16, row-major, stride LSTR ----
    if (AF16) {
      const h16* Ap = (const h16*)Aall + i1 * sA1 + i2 * sA2;
      for (int i = tid; i < BM * BK / 8; i += 256) {
        int r = i >> 2, c8 = (i & 3) * 8;
        int gr = m0 + r, gc = k0 + c8;
        v8h vv;
        if (gr < M && gc < K) vv = *(const v8h*)(Ap + (long long)gr * lda + gc);
        else { for (int e = 0; e < 8; ++e) vv[e] = (h16)0.0f; }
        *(v8h*)&sA[r * LSTR + c8] = vv;
      }
      if (kt + 1 < nk) { int gr = m0 + (tid & 127);
        if (gr < M) __builtin_prefetch(Ap + (long long)gr * lda + k0 + BK, 0, 1); }
    } else {
      const float* Ap = (const float*)Aall + i1 * sA1 + i2 * sA2;
      for (int i = tid; i < BM * BK / 4; i += 256) {
        int r = i >> 3, c4 = (i & 7) * 4;
        int gr = m0 + r, gc = k0 + c4;
        v4f vv = {0.f, 0.f, 0.f, 0.f};
        if (gr < M && gc < K) vv = *(const v4f*)(Ap + (long long)gr * lda + gc);
        h16* dst = &sA[r * LSTR + c4];
        dst[0] = (h16)vv[0]; dst[1] = (h16)vv[1]; dst[2] = (h16)vv[2]; dst[3] = (h16)vv[3];
      }
      if (kt + 1 < nk) { int gr = m0 + (tid & 127);
        if (gr < M) __builtin_prefetch(Ap + (long long)gr * lda + k0 + BK, 0, 1); }
    }

    // ---- stage B tile transposed: sB[n][k] ----
    if (TRANSB) { // B given as [N,K] row-major -> straight copy
      for (int i = tid; i < BN * BK / 4; i += 256) {
        int n = i >> 3, c4 = (i & 7) * 4;
        int gn = n0 + n, gc = k0 + c4;
        v4f vv = {0.f, 0.f, 0.f, 0.f};
        if (gn < N && gc < K) vv = *(const v4f*)(Bp + (long long)gn * ldb + gc);
        h16* dst = &sB[n * LSTR + c4];
        dst[0] = (h16)vv[0]; dst[1] = (h16)vv[1]; dst[2] = (h16)vv[2]; dst[3] = (h16)vv[3];
      }
    } else {      // B given as [K,N] row-major -> transpose while staging
      for (int i = tid; i < BN * BK; i += 256) {
        int n = i & (BN - 1), k = i >> 6;
        int gn = n0 + n, gk = k0 + k;
        float v = (gn < N && gk < K) ? Bp[(long long)gk * ldb + gn] : 0.0f;
        sB[n * LSTR + k] = (h16)v;
      }
    }
    __syncthreads();

    // ---- fragments per ISA VGPR layouts (wave32) ----
    v16h af[2], bf[2];
#pragma unroll
    for (int mt = 0; mt < 2; ++mt) {
      int r = wm * 32 + mt * 16 + l15;
      int ko = lhi * 8; // lanes 0-15: K 0-7 & 16-23; lanes 16-31: K 8-15 & 24-31
      v8h lo = *(const v8h*)&sA[r * LSTR + ko];
      v8h hi = *(const v8h*)&sA[r * LSTR + 16 + ko];
      af[mt] = __builtin_shufflevector(lo, hi, 0,1,2,3,4,5,6,7,8,9,10,11,12,13,14,15);
    }
#pragma unroll
    for (int nt = 0; nt < 2; ++nt) {
      int c  = wn * 32 + nt * 16 + l15;
      int kb = lhi * 8; // lanes 0-15: K 0-7 & 16-23; lanes 16-31: K 8-15 & 24-31
      v8h lo = *(const v8h*)&sB[c * LSTR + kb];
      v8h hi = *(const v8h*)&sB[c * LSTR + kb + 16];
      bf[nt] = __builtin_shufflevector(lo, hi, 0,1,2,3,4,5,6,7,8,9,10,11,12,13,14,15);
    }
#pragma unroll
    for (int mt = 0; mt < 2; ++mt)
#pragma unroll
      for (int nt = 0; nt < 2; ++nt)
        acc[mt][nt] = __builtin_amdgcn_wmma_f32_16x16x32_f16(
            false, af[mt], false, bf[nt], (short)0, acc[mt][nt], false, false);
    __syncthreads();
  }

  // ---- epilogue: C layout VGPR r -> M = r (lanes 0-15) / r+8 (lanes 16-31) ----
#pragma unroll
  for (int mt = 0; mt < 2; ++mt)
#pragma unroll
    for (int nt = 0; nt < 2; ++nt)
#pragma unroll
      for (int r8 = 0; r8 < 8; ++r8) {
        int row = m0 + wm * 32 + mt * 16 + lhi * 8 + r8;
        int col = n0 + wn * 32 + nt * 16 + l15;
        if (row < M && col < N) {
          float v = acc[mt][nt][r8];
          if (bias) v += bias[col];
          if (RELU) v = fmaxf(v, 0.0f);
          long long idx = cOff + (long long)row * ldc + col;
          if (OUTF16) ((h16*)Call)[idx] = (h16)v;
          else        ((float*)Call)[idx] = v;
        }
      }
}

// ============================================================================
// Small support kernels
// ============================================================================
__global__ void k_compound_concat(const int* E, const int* C, const float* P3,
                                  const float* elemEmb, const float* chgEmb,
                                  const float* posW, const float* posB,
                                  float* out, int rows)
{
  int idx = blockIdx.x * blockDim.x + threadIdx.x;
  int total = rows * 3 * cfg::D;
  if (idx >= total) return;
  int row = idx / (3 * cfg::D), c = idx % (3 * cfg::D);
  float v;
  if (c < cfg::D)            v = elemEmb[E[row] * cfg::D + c];
  else if (c < 2 * cfg::D)   v = chgEmb[C[row] * cfg::D + (c - cfg::D)];
  else {
    int d = c - 2 * cfg::D;
    v = fmaf(P3[row*3+0], posW[d],
        fmaf(P3[row*3+1], posW[cfg::D + d],
        fmaf(P3[row*3+2], posW[2*cfg::D + d], posB[d])));
    v = fmaxf(v, 0.0f);
  }
  out[idx] = v;
}

__global__ void k_theo_concat(const int* E, const float* P3, const float* elemEmb,
                              const float* posW, const float* posB, float* out, int rows)
{
  int idx = blockIdx.x * blockDim.x + threadIdx.x;
  int total = rows * 2 * cfg::D;
  if (idx >= total) return;
  int row = idx / (2 * cfg::D), c = idx % (2 * cfg::D);
  float v;
  if (c < cfg::D) v = elemEmb[E[row] * cfg::D + c];
  else {
    int d = c - cfg::D;
    v = fmaf(P3[row*3+0], posW[d],
        fmaf(P3[row*3+1], posW[cfg::D + d],
        fmaf(P3[row*3+2], posW[2*cfg::D + d], posB[d])));
    v = fmaxf(v, 0.0f);
  }
  out[idx] = v;
}

__global__ void k_copy2(const float* a, const float* b, float* out, int rows, int d)
{
  int idx = blockIdx.x * blockDim.x + threadIdx.x;
  if (idx >= rows * 2 * d) return;
  int row = idx / (2 * d), c = idx % (2 * d);
  out[idx] = (c < d) ? a[row * d + c] : b[row * d + (c - d)];
}

__global__ void k_rowsum(const float* emb, float* out, int nrows, int d)
{
  int r = threadIdx.x;
  if (r < nrows) { float s = 0.f; for (int i = 0; i < d; ++i) s += emb[r * d + i]; out[r] = s; }
}

// softmax over last dim of scores [z][Sq][Skv]; z = h*B+b.
// val = s*scale, optionally *= bsum[adj[b,q,k]], optionally causal mask (k>q -> -inf)
__global__ __launch_bounds__(128)
void k_softmax(float* S, int Sq, int Skv, float scale,
               const int* adj, const float* bsum, int Bdim, int causal)
{
  int q = blockIdx.x, z = blockIdx.y, tid = threadIdx.x;
  int b = z % Bdim;
  float* row = S + ((long long)z * Sq + q) * Skv;
  const int* arow = adj ? (adj + ((long long)b * Sq + q) * Skv) : nullptr;
  __shared__ float red[128];
  float vals[4];
  int nper = (Skv + 127) / 128;
  float mx = -INFINITY;
  for (int j = 0; j < nper; ++j) {
    int k = tid + j * 128;
    float v = -INFINITY;
    if (k < Skv) {
      v = row[k] * scale;
      if (arow) v *= bsum[arow[k]];
      if (causal && k > q) v = -INFINITY;
    }
    vals[j] = v; mx = fmaxf(mx, v);
  }
  red[tid] = mx; __syncthreads();
  for (int s = 64; s; s >>= 1) { if (tid < s) red[tid] = fmaxf(red[tid], red[tid + s]); __syncthreads(); }
  mx = red[0]; __syncthreads();
  float sum = 0.f;
  for (int j = 0; j < nper; ++j) {
    float e = (vals[j] == -INFINITY) ? 0.f : __expf(vals[j] - mx);
    vals[j] = e; sum += e;
  }
  red[tid] = sum; __syncthreads();
  for (int s = 64; s; s >>= 1) { if (tid < s) red[tid] += red[tid + s]; __syncthreads(); }
  float inv = 1.f / red[0];
  for (int j = 0; j < nper; ++j) { int k = tid + j * 128; if (k < Skv) row[k] = vals[j] * inv; }
}

// x = LayerNorm(a + x) * g + beta   (in place on x), D=128, one row per block
__global__ __launch_bounds__(128)
void k_add_ln(const float* a, float* x, const float* g, const float* beta)
{
  int row = blockIdx.x, tid = threadIdx.x;
  float v = a[row * 128 + tid] + x[row * 128 + tid];
  __shared__ float red[128];
  red[tid] = v; __syncthreads();
  for (int s = 64; s; s >>= 1) { if (tid < s) red[tid] += red[tid + s]; __syncthreads(); }
  float mean = red[0] * (1.f / 128.f); __syncthreads();
  float d = v - mean;
  red[tid] = d * d; __syncthreads();
  for (int s = 64; s; s >>= 1) { if (tid < s) red[tid] += red[tid + s]; __syncthreads(); }
  float var = red[0] * (1.f / 128.f);
  x[row * 128 + tid] = d * rsqrtf(var + 1e-5f) * g[tid] + beta[tid];
}

// legacy softmax over batch axis (B=2) of [B,T,E]
__global__ void k_elem_softmax(const float* lg, float* out, int T_, int E_)
{
  int idx = blockIdx.x * blockDim.x + threadIdx.x;
  if (idx >= T_ * E_) return;
  int t = idx / E_, e = idx % E_;
  float l0 = lg[(long long)t * E_ + e];
  float l1 = lg[((long long)T_ + t) * E_ + e];
  float m = fmaxf(l0, l1);
  float e0 = __expf(l0 - m), e1 = __expf(l1 - m);
  float inv = 1.f / (e0 + e1);
  out[(long long)t * E_ + e] = e0 * inv;
  out[((long long)T_ + t) * E_ + e] = e1 * inv;
}

// softmax over dim1 (first T) of [B,T,T,NB]
__global__ void k_bond_softmax(const float* lg, float* out)
{
  using namespace cfg;
  int idx = blockIdx.x * blockDim.x + threadIdx.x;
  if (idx >= B * T * NB) return;
  int b = idx / (T * NB);
  int rem = idx - b * T * NB;
  int t2 = rem / NB, nb = rem % NB;
  long long base = (long long)b * T * T * NB + (long long)t2 * NB + nb;
  long long stride = (long long)T * NB;
  float mx = -INFINITY;
  for (int t1 = 0; t1 < T; ++t1) mx = fmaxf(mx, lg[base + t1 * stride]);
  float s = 0.f;
  for (int t1 = 0; t1 < T; ++t1) s += __expf(lg[base + t1 * stride] - mx);
  float inv = 1.f / s;
  for (int t1 = 0; t1 < T; ++t1) out[base + t1 * stride] = __expf(lg[base + t1 * stride] - mx) * inv;
}

// ============================================================================
// Host side
// ============================================================================
static void gemm(hipStream_t st, const void* A, const float* Bm, const float* bias, void* C,
                 int M, int N, int K, int lda, int ldb, int ldc,
                 int nb1, int nb2,
                 long long sA1, long long sA2, long long sB1, long long sB2,
                 long long sC1, long long sC2, long long sBias1,
                 bool transB, bool relu, bool aF16, bool outF16)
{
  dim3 g((N + 63) / 64, (M + 127) / 128, nb1 * nb2), blk(256);
#define GEMM_CASE(TB, RL, AF, OF)                                                      \
  if (transB == TB && relu == RL && aF16 == AF && outF16 == OF) {                      \
    gemm_wmma<TB, RL, AF, OF><<<g, blk, 0, st>>>(A, Bm, bias, C, M, N, K, lda, ldb,    \
        ldc, nb2, sA1, sA2, sB1, sB2, sC1, sC2, sBias1);                               \
    return; }
  GEMM_CASE(false, false, false, false)
  GEMM_CASE(false, true,  false, false)
  GEMM_CASE(false, true,  false, true )
  GEMM_CASE(false, true,  true,  true )
  GEMM_CASE(false, false, true,  false)
  GEMM_CASE(true,  false, false, false)
#undef GEMM_CASE
}

static void gemm1(hipStream_t st, const void* A, const float* Bm, const float* bias, void* C,
                  int M, int N, int K, int lda, int ldb, int ldc,
                  bool transB, bool relu, bool aF16, bool outF16)
{ gemm(st, A, Bm, bias, C, M, N, K, lda, ldb, ldc, 1, 1, 0,0,0,0,0,0,0, transB, relu, aF16, outF16); }

struct MhaW { const float *wq,*wqb,*wk,*wkb,*wv,*wvb,*ow,*ob; };

static void run_mha(hipStream_t st, const float* Xq, const float* Xkv,
                    int Sq, int Skv, const MhaW& w, float* out,
                    const int* adj, const float* bsum, bool causal,
                    float* qb, float* kb, float* vb, float* sb, float* cat)
{
  using namespace cfg;
  long long BSqD = (long long)B * Sq * D, BSkD = (long long)B * Skv * D;
  // per-head projections (batched over H)
  gemm(st, Xq,  w.wq, w.wqb, qb, B*Sq,  D, D, D, D, D, H, 1, 0,0, (long long)D*D,0, BSqD,0, D, false,false,false,false);
  gemm(st, Xkv, w.wk, w.wkb, kb, B*Skv, D, D, D, D, D, H, 1, 0,0, (long long)D*D,0, BSkD,0, D, false,false,false,false);
  gemm(st, Xkv, w.wv, w.wvb, vb, B*Skv, D, D, D, D, D, H, 1, 0,0, (long long)D*D,0, BSkD,0, D, false,false,false,false);
  // scores = Q @ K^T  (B operand transposed -> straight-copy staging)
  gemm(st, qb, kb, nullptr, sb, Sq, Skv, D, D, D, Skv, H, B,
       BSqD, (long long)Sq*D, BSkD, (long long)Skv*D,
       (long long)B*Sq*Skv, (long long)Sq*Skv, 0, true, false, false, false);
  // softmax (+ graph weights + causal mask), in place
  k_softmax<<<dim3(Sq, H*B), dim3(128), 0, st>>>(sb, Sq, Skv, 1.0f / sqrtf((float)Skv),
                                                 adj, bsum, B, causal ? 1 : 0);
  // O = P @ V, written into concat buffer [b][q][h][e]
  gemm(st, sb, vb, nullptr, cat, Sq, D, Skv, Skv, D, H*D, H, B,
       (long long)B*Sq*Skv, (long long)Sq*Skv, BSkD, (long long)Skv*D,
       (long long)D, (long long)Sq*H*D, 0, false, false, false, false);
  // output projection
  gemm1(st, cat, w.ow, w.ob, out, B*Sq, D, H*D, H*D, D, D, false, false, false, false);
}

// params flattened after the 11 tensor inputs, in _make_params() insertion order
enum {
  I_RE = 0, I_RC, I_RP, I_RADJ, I_PE, I_PC, I_PP, I_PADJ, I_TE, I_TP, I_TADJ,
  R_ELEM, R_CHG, R_BOND, R_POSW, R_POSB, R_FFW, R_FFB,
  R_WQ, R_WQB, R_WK, R_WKB, R_WV, R_WVB, R_OW, R_OB,
  Q_ELEM, Q_CHG, Q_BOND, Q_POSW, Q_POSB, Q_FFW, Q_FFB,
  Q_WQ, Q_WQB, Q_WK, Q_WKB, Q_WV, Q_WVB, Q_OW, Q_OB,
  T_ELEM, T_BOND, T_POSW, T_POSB, T_FFW, T_FFB,
  T_WQ, T_WQB, T_WK, T_WKB, T_WV, T_WVB, T_OW, T_OB,
  COMP_W, COMP_B,
  E_WQ, E_WQB, E_WK, E_WKB, E_WV, E_WVB, E_OW, E_OB,
  DS_WQ, DS_WQB, DS_WK, DS_WKB, DS_WV, DS_WVB, DS_OW, DS_OB,
  DC_WQ, DC_WQB, DC_WK, DC_WKB, DC_WV, DC_WVB, DC_OW, DC_OB,
  E_FF1W, E_FF1B, E_FF2W, E_FF2B, E_LNG, E_LNB,
  D_FF1W, D_FF1B, D_FF2W, D_FF2B, D_LNG, D_LNB,
  TR_E1W, TR_E1B, TR_E2W, TR_E2B,
  TR_P1W, TR_P1B, TR_P2W, TR_P2B,
  TR_B1W, TR_B1B, TR_B2AW, TR_B2AB, TR_B2BW, TR_B2BB
};

extern "C" void kernel_launch(void* const* d_in, const int* in_sizes, int n_in,
                              void* d_out, int out_size, void* d_ws, size_t ws_size,
                              hipStream_t stream)
{
  using namespace cfg;
  (void)in_sizes; (void)n_in; (void)out_size; (void)ws_size;

  auto Pf = [&](int i) { return (const float*)d_in[i]; };
  auto Pi = [&](int i) { return (const int*)d_in[i]; };

  const int rowsC = B * SC;  // 512
  const int rowsT = B * T;   // 768

  // ---- workspace bump allocator ----
  char* wsb = (char*)d_ws; size_t off = 0;
  auto alloc = [&](size_t bytes) -> void* {
    size_t a = (off + 255) & ~(size_t)255; off = a + bytes; return wsb + a;
  };
  float* catin = (float*)alloc((size_t)rowsC * 3 * D * 4);   // 512x384 (>= 768x256 too)
  float* gbuf  = (float*)alloc((size_t)rowsT * D * 4);
  float* rx    = (float*)alloc((size_t)rowsC * D * 4);
  float* px    = (float*)alloc((size_t)rowsC * D * 4);
  float* xenc  = (float*)alloc((size_t)rowsC * D * 4);
  float* txb   = (float*)alloc((size_t)rowsT * D * 4);
  float* attn  = (float*)alloc((size_t)rowsT * D * 4);
  float* ffb   = (float*)alloc((size_t)rowsT * F * 4);
  float* qb    = (float*)alloc((size_t)H * rowsT * D * 4);
  float* kb    = (float*)alloc((size_t)H * rowsT * D * 4);
  float* vb    = (float*)alloc((size_t)H * rowsT * D * 4);
  float* sb    = (float*)alloc((size_t)H * B * T * T * 4);
  float* cat   = (float*)alloc((size_t)rowsT * H * D * 4);
  float* bsumR = (float*)alloc(NB * 4);
  float* bsumP = (float*)alloc(NB * 4);
  float* bsumT = (float*)alloc(NB * 4);
  float* ellog = (float*)alloc((size_t)rowsT * NELEM * 4);
  h16*   up    = (h16*)alloc((size_t)rowsT * T * D * 2);     // 75.5 MB (f16)
  h16*   h2    = (h16*)alloc((size_t)rowsT * T * D * 2);     // 75.5 MB (f16)
  float* blog  = (float*)alloc((size_t)rowsT * T * NB * 4);  // 9.4 MB

  // ---- bond-embedding row sums (graph-attention weight = s * bsum[adj]) ----
  k_rowsum<<<1, 32, 0, stream>>>(Pf(R_BOND), bsumR, NB, D);
  k_rowsum<<<1, 32, 0, stream>>>(Pf(Q_BOND), bsumP, NB, D);
  k_rowsum<<<1, 32, 0, stream>>>(Pf(T_BOND), bsumT, NB, D);

  // ---- reactant compound ----
  {
    int tot = rowsC * 3 * D;
    k_compound_concat<<<(tot + 255) / 256, 256, 0, stream>>>(
        Pi(I_RE), Pi(I_RC), Pf(I_RP), Pf(R_ELEM), Pf(R_CHG), Pf(R_POSW), Pf(R_POSB), catin, rowsC);
    gemm1(stream, catin, Pf(R_FFW), Pf(R_FFB), gbuf, rowsC, D, 3*D, 3*D, D, D, false, true, false, false);
    MhaW w{Pf(R_WQ), Pf(R_WQB), Pf(R_WK), Pf(R_WKB), Pf(R_WV), Pf(R_WVB), Pf(R_OW), Pf(R_OB)};
    run_mha(stream, gbuf, gbuf, SC, SC, w, rx, Pi(I_RADJ), bsumR, false, qb, kb, vb, sb, cat);
  }
  // ---- product compound ----
  {
    int tot = rowsC * 3 * D;
    k_compound_concat<<<(tot + 255) / 256, 256, 0, stream>>>(
        Pi(I_PE), Pi(I_PC), Pf(I_PP), Pf(Q_ELEM), Pf(Q_CHG), Pf(Q_POSW), Pf(Q_POSB), catin, rowsC);
    gemm1(stream, catin, Pf(Q_FFW), Pf(Q_FFB), gbuf, rowsC, D, 3*D, 3*D, D, D, false, true, false, false);
    MhaW w{Pf(Q_WQ), Pf(Q_WQB), Pf(Q_WK), Pf(Q_WKB), Pf(Q_WV), Pf(Q_WVB), Pf(Q_OW), Pf(Q_OB)};
    run_mha(stream, gbuf, gbuf, SC, SC, w, px, Pi(I_PADJ), bsumP, false, qb, kb, vb, sb, cat);
  }
  // ---- combine compounds ----
  {
    int tot = rowsC * 2 * D;
    k_copy2<<<(tot + 255) / 256, 256, 0, stream>>>(rx, px, catin, rowsC, D);
    gemm1(stream, catin, Pf(COMP_W), Pf(COMP_B), xenc, rowsC, D, 2*D, 2*D, D, D, false, true, false, false);
  }
  // ---- theozyme graph attention (causal + bond weights) ----
  {
    int tot = rowsT * 2 * D;
    k_theo_concat<<<(tot + 255) / 256, 256, 0, stream>>>(
        Pi(I_TE), Pf(I_TP), Pf(T_ELEM), Pf(T_POSW), Pf(T_POSB), catin, rowsT);
    gemm1(stream, catin, Pf(T_FFW), Pf(T_FFB), gbuf, rowsT, D, 2*D, 2*D, D, D, false, true, false, false);
    MhaW w{Pf(T_WQ), Pf(T_WQB), Pf(T_WK), Pf(T_WKB), Pf(T_WV), Pf(T_WVB), Pf(T_OW), Pf(T_OB)};
    run_mha(stream, gbuf, gbuf, T, T, w, txb, Pi(I_TADJ), bsumT, true, qb, kb, vb, sb, cat);
  }
  // ---- encoder ----
  for (int i = 0; i < L; ++i) {
    MhaW w{Pf(E_WQ) + (size_t)i*H*D*D, Pf(E_WQB) + (size_t)i*H*D,
           Pf(E_WK) + (size_t)i*H*D*D, Pf(E_WKB) + (size_t)i*H*D,
           Pf(E_WV) + (size_t)i*H*D*D, Pf(E_WVB) + (size_t)i*H*D,
           Pf(E_OW) + (size_t)i*H*D*D, Pf(E_OB) + (size_t)i*D};
    run_mha(stream, xenc, xenc, SC, SC, w, attn, nullptr, nullptr, false, qb, kb, vb, sb, cat);
    k_add_ln<<<rowsC, 128, 0, stream>>>(attn, xenc, Pf(E_LNG) + i*D, Pf(E_LNB) + i*D);
    gemm1(stream, xenc, Pf(E_FF1W) + (size_t)i*D*F, Pf(E_FF1B) + (size_t)i*F, ffb, rowsC, F, D, D, F, F, false, true, false, false);
    gemm1(stream, ffb, Pf(E_FF2W) + (size_t)i*F*D, Pf(E_FF2B) + (size_t)i*D, attn, rowsC, D, F, F, D, D, false, false, false, false);
    k_add_ln<<<rowsC, 128, 0, stream>>>(attn, xenc, Pf(E_LNG) + i*D, Pf(E_LNB) + i*D);
  }
  // ---- decoder ----
  for (int i = 0; i < L; ++i) {
    MhaW ws_{Pf(DS_WQ) + (size_t)i*H*D*D, Pf(DS_WQB) + (size_t)i*H*D,
             Pf(DS_WK) + (size_t)i*H*D*D, Pf(DS_WKB) + (size_t)i*H*D,
             Pf(DS_WV) + (size_t)i*H*D*D, Pf(DS_WVB) + (size_t)i*H*D,
             Pf(DS_OW) + (size_t)i*H*D*D, Pf(DS_OB) + (size_t)i*D};
    run_mha(stream, txb, txb, T, T, ws_, attn, nullptr, nullptr, true, qb, kb, vb, sb, cat);
    k_add_ln<<<rowsT, 128, 0, stream>>>(attn, txb, Pf(D_LNG) + i*D, Pf(D_LNB) + i*D);
    MhaW wc{Pf(DC_WQ) + (size_t)i*H*D*D, Pf(DC_WQB) + (size_t)i*H*D,
            Pf(DC_WK) + (size_t)i*H*D*D, Pf(DC_WKB) + (size_t)i*H*D,
            Pf(DC_WV) + (size_t)i*H*D*D, Pf(DC_WVB) + (size_t)i*H*D,
            Pf(DC_OW) + (size_t)i*H*D*D, Pf(DC_OB) + (size_t)i*D};
    run_mha(stream, txb, xenc, T, SC, wc, attn, nullptr, nullptr, false, qb, kb, vb, sb, cat);
    k_add_ln<<<rowsT, 128, 0, stream>>>(attn, txb, Pf(D_LNG) + i*D, Pf(D_LNB) + i*D);
    gemm1(stream, txb, Pf(D_FF1W) + (size_t)i*D*F, Pf(D_FF1B) + (size_t)i*F, ffb, rowsT, F, D, D, F, F, false, true, false, false);
    gemm1(stream, ffb, Pf(D_FF2W) + (size_t)i*F*D, Pf(D_FF2B) + (size_t)i*D, attn, rowsT, D, F, F, D, D, false, false, false, false);
    k_add_ln<<<rowsT, 128, 0, stream>>>(attn, txb, Pf(D_LNG) + i*D, Pf(D_LNB) + i*D);
  }
  // ---- transducer heads: outputs (elem, pos, bond) concatenated in d_out ----
  float* elemOut = (float*)d_out;
  float* posOut  = elemOut + (size_t)B * T * NELEM;
  float* bondOut = posOut + (size_t)B * T * 3;

  // elem: relu(tx@e1+b)@e2+b -> softmax over batch axis
  gemm1(stream, txb, Pf(TR_E1W), Pf(TR_E1B), ffb, rowsT, D, D, D, D, D, false, true, false, false);
  gemm1(stream, ffb, Pf(TR_E2W), Pf(TR_E2B), ellog, rowsT, NELEM, D, D, NELEM, NELEM, false, false, false, false);
  k_elem_softmax<<<(T * NELEM + 255) / 256, 256, 0, stream>>>(ellog, elemOut, T, NELEM);

  // pos: relu(tx@p1+b)@p2+b
  gemm1(stream, txb, Pf(TR_P1W), Pf(TR_P1B), ffb, rowsT, D, D, D, D, D, false, true, false, false);
  gemm1(stream, ffb, Pf(TR_P2W), Pf(TR_P2B), posOut, rowsT, 3, D, D, 3, 3, false, false, false, false);

  // bond: up = relu(tx @ [128 x T*D]) (f16), h2 = relu(up @ b2a) (f16),
  //       logits = h2 @ b2b + b, softmax over dim1
  gemm1(stream, txb, Pf(TR_B1W), Pf(TR_B1B), up, rowsT, T * D, D, D, T * D, T * D, false, true, false, true);
  gemm1(stream, up, Pf(TR_B2AW), Pf(TR_B2AB), h2, rowsT * T, D, D, D, D, D, false, true, true, true);
  gemm1(stream, h2, Pf(TR_B2BW), Pf(TR_B2BB), blog, rowsT * T, NB, D, D, NB, NB, false, false, true, false);
  k_bond_softmax<<<(B * T * NB + 255) / 256, 256, 0, stream>>>(blog, bondOut);
}
